// WayfinderAttentionMLX_30915174597244
// MI455X (gfx1250) — matrix operand order repaired and checked by reading, not compiled, not run on hardware
//
#include <hip/hip_runtime.h>
#include <hip/hip_bf16.h>

typedef __attribute__((ext_vector_type(16))) _Float16 v16h;
typedef __attribute__((ext_vector_type(8)))  _Float16 v8h;
typedef __attribute__((ext_vector_type(2)))  _Float16 v2h;
typedef __attribute__((ext_vector_type(8)))  float    v8f;

#define H_ 16
#define T_ 1024
#define DH_ 64
#define D_ 64
#define NEG_INF_ (-1e30f)
#define LOG2E_ 1.4426950408889634f

// ---------------------------------------------------------------------------
// Kernel 1: one streaming pass converting q (pre-scaled by log2e/sqrt(dh)),
// k, v from f32 to f16 scratch. Halves the random-gather traffic (the
// dominant cost) and produces WMMA-ready operands. The log2e factor turns
// the softmax into a bare exp2 (v_exp_f32) with no per-element multiply.
// ---------------------------------------------------------------------------
__global__ void __launch_bounds__(256)
wayfinder_cvt_kernel(const float* __restrict__ q,
                     const float* __restrict__ k,
                     const float* __restrict__ v,
                     _Float16* __restrict__ q16,
                     _Float16* __restrict__ k16,
                     _Float16* __restrict__ v16,
                     int n) {
  int i = blockIdx.x * blockDim.x + threadIdx.x;
  if (i < n) {
    q16[i] = (_Float16)(q[i] * (0.125f * LOG2E_));   // 1/sqrt(64) * log2(e)
    k16[i] = (_Float16)k[i];
    v16[i] = (_Float16)v[i];
  }
}

static __device__ inline v16h cat8(v8h lo, v8h hi) {
  v16h r;
#pragma unroll
  for (int i = 0; i < 8; ++i) { r[i] = lo[i]; r[8 + i] = hi[i]; }
  return r;
}

static __device__ inline float readlane_f32(float x, int lane) {
  return __int_as_float(__builtin_amdgcn_readlane(__float_as_int(x), lane));
}

// ---------------------------------------------------------------------------
// Kernel 2: one wave32 per query.
//   Scores: v_wmma_f32_16x16x32_f16, A = gathered K tile (16 nbr x 32 dim),
//           B = q broadcast to all 16 columns, C preloaded with
//           badd[j] = mask ? bias*log2e : -1e30  =>  D = log2-domain score.
//   Softmax: branchless base-2; masked entries underflow in exp2; all-masked
//           row killed by a wave-uniform `live` factor folded into 1/denom,
//           which is applied once to the output accumulators.
//   Output: coalesced f16 V gather; weights/indices broadcast via v_readlane.
// ---------------------------------------------------------------------------
__global__ void __launch_bounds__(256)
wayfinder_attn_kernel(const _Float16* __restrict__ q16,
                      const _Float16* __restrict__ k16,
                      const _Float16* __restrict__ v16,
                      const float* __restrict__ ebias,
                      const int* __restrict__ nidx,
                      const int* __restrict__ etype,
                      float* __restrict__ out) {
  __shared__ int   s_idx [8][D_];
  __shared__ float s_badd[8][D_];

  const int wv   = threadIdx.x >> 5;
  const int lane = threadIdx.x & 31;
  const int qid  = blockIdx.x * 8 + wv;       // 0 .. H*T-1
  const int h    = qid >> 10;
  const int t    = qid & (T_ - 1);
  const int half = lane >> 4;
  const int m    = lane & 15;

  // ---- Phase A: neighbor metadata (coalesced); keep clamped idx in regs ----
  const int ibase = (h * T_ + t) * D_;
  int ivr[2];
#pragma unroll
  for (int rep = 0; rep < 2; ++rep) {
    int j  = lane + rep * 32;
    int iv = nidx[ibase + j];
    int et = etype[ibase + j];
    int sc = iv < 0 ? 0 : (iv > (T_ - 1) ? (T_ - 1) : iv);
    // unconditional clamped load, then branchless selects
    float bv   = ebias[(et > 0) ? (et - 1) : 0] * LOG2E_;
    bv         = (et == 0) ? 0.0f : bv;
    float badd = (iv >= 0 && iv <= t) ? bv : NEG_INF_;
    ivr[rep] = sc;
    s_idx [wv][j] = sc;
    s_badd[wv][j] = badd;
  }
  asm volatile("s_wait_dscnt 0x0" ::: "memory");   // same-wave LDS handoff

  // ---- Phase B: scores via WMMA, bias/mask folded into C ----
  const _Float16* qrow = q16 + (size_t)(h * T_ + t) * DH_;
  v16h b0 = cat8(*(const v8h*)(qrow + half * 8), *(const v8h*)(qrow + 16 + half * 8));        // dims  0..31 chunk
  v16h b1 = cat8(*(const v8h*)(qrow + 32 + half * 8), *(const v8h*)(qrow + 48 + half * 8));   // dims 32..63 chunk

  const _Float16* kh = k16 + (size_t)h * T_ * DH_;
  v8f acc[4];
#pragma unroll
  for (int g = 0; g < 4; ++g) {
    int sidx = s_idx[wv][g * 16 + m];                // neighbor 16g+m
    const _Float16* krow = kh + (unsigned)(sidx * DH_);
    const int c0 = half * 8;                         // A-layout K split
    v8h a00 = *(const v8h*)(krow + c0);              // dims c0   .. c0+7
    v8h a01 = *(const v8h*)(krow + c0 + 16);         // dims c0+16.. c0+23
    v8h a10 = *(const v8h*)(krow + c0 + 32);
    v8h a11 = *(const v8h*)(krow + c0 + 48);
    v8f c;
#pragma unroll
    for (int r = 0; r < 8; ++r)                      // C init: bias-or--1e30
      c[r] = s_badd[wv][g * 16 + half * 8 + r];      // 2-addr LDS broadcast
    c = __builtin_amdgcn_wmma_f32_16x16x32_f16(false, cat8(a00, a01),
                                               false, b0, (short)0, c,
                                               false, false);
    c = __builtin_amdgcn_wmma_f32_16x16x32_f16(false, cat8(a10, a11),
                                               false, b1, (short)0, c,
                                               false, false);
    acc[g] = c;   // lane holds log2-score[16g + 8*half + r] in elem r
  }

  // ---- Phase C: branchless base-2 softmax over 64 scores ----
  float mx = NEG_INF_;
#pragma unroll
  for (int g = 0; g < 4; ++g)
#pragma unroll
    for (int r = 0; r < 8; ++r) mx = fmaxf(mx, acc[g][r]);
  mx = fmaxf(mx, __shfl_xor(mx, 16, 32));

  float sv[4][8];
  float psum = 0.f;
#pragma unroll
  for (int g = 0; g < 4; ++g)
#pragma unroll
    for (int r = 0; r < 8; ++r) {
      float e = __builtin_amdgcn_exp2f(acc[g][r] - mx);  // masked -> 0
      sv[g][r] = e;
      psum += e;
    }
  float denom = psum + __shfl_xor(psum, 16, 32);
  float live  = (mx > -0.5e30f) ? 1.0f : 0.0f;       // all-masked row -> 0
  float inv   = live / fmaxf(denom, 1e-9f);          // applied to y at end

  // ---- Phase D: y = inv * sum_j e_j * V[idx_j]; lane owns dims 2L,2L+1 ----
  // sj via readlane of the Phase-A index registers (uniform -> cheap addr),
  // e_j via readlane of the replicated weight registers. Zero LDS traffic.
  const _Float16* vh = v16 + (size_t)h * T_ * DH_;
  const int cpos = 2 * lane;
  float y0 = 0.f, y1 = 0.f;
#pragma unroll
  for (int j = 0; j < D_; ++j) {
    const int o = j & 15;
    int   sj = __builtin_amdgcn_readlane(ivr[j >> 5], j & 31);
    float wj = readlane_f32(sv[j >> 4][o & 7], (o < 8) ? 0 : 16);
    v2h vv = *(const v2h*)(vh + (unsigned)(sj * DH_ + cpos));  // 128B/row coalesced
    y0 += wj * (float)vv[0];
    y1 += wj * (float)vv[1];
  }
  float2 yo = make_float2(y0 * inv, y1 * inv);
  *(float2*)(out + (size_t)(h * T_ + t) * DH_ + cpos) = yo;
}

extern "C" void kernel_launch(void* const* d_in, const int* in_sizes, int n_in,
                              void* d_out, int out_size, void* d_ws, size_t ws_size,
                              hipStream_t stream) {
  const float* q     = (const float*)d_in[0];   // (1,16,1024,64) f32
  const float* k     = (const float*)d_in[1];   // (1,16,1024,64) f32
  const float* v     = (const float*)d_in[2];   // (1,16,1024,64) f32
  const float* ebias = (const float*)d_in[3];   // (4,) f32
  const int*   nidx  = (const int*)d_in[4];     // (16,1024,64) i32
  const int*   etype = (const int*)d_in[5];     // (16,1024,64) i32

  const int n = H_ * T_ * DH_;                  // 1,048,576 per tensor
  _Float16* k16 = (_Float16*)d_ws;              // 2 MB
  _Float16* v16 = k16 + n;                      // 2 MB
  _Float16* q16 = v16 + n;                      // 2 MB  (6 MB scratch total)

  wayfinder_cvt_kernel<<<(n + 255) / 256, 256, 0, stream>>>(
      q, k, v, q16, k16, v16, n);

  const int nq = H_ * T_;                       // 16384 queries, 8 waves/block
  wayfinder_attn_kernel<<<nq / 8, 256, 0, stream>>>(
      q16, k16, v16, ebias, nidx, etype, (float*)d_out);
}
